// NeuralEmbeddingLayer_45329084842551
// MI455X (gfx1250) — compile-verified, matrix operand order checked by reading, not run on hardware
//
#include <hip/hip_runtime.h>
#include <hip/hip_bf16.h>
#include <math.h>

#define BB      16
#define TT      512
#define CC      768
#define DIN     3072
#define KSTACK  (4 * DIN)      /* 12288 */
#define HID     1024
#define LOUT    255
#define SEQ     257
#define NDATES  16

typedef __attribute__((ext_vector_type(16))) __bf16 v16bf;
typedef __attribute__((ext_vector_type(8)))  __bf16 v8bf;
typedef __attribute__((ext_vector_type(8)))  float  v8f;

// Per-lane 16-element bf16 WMMA fragment (16-bit A/B 16x32 layout): lane half
// h holds K = {h*8..h*8+7} and {16+h*8..16+h*8+7}; caller pre-adds h*8 so the
// two 8-element chunks sit at +0 and +16 elements.
static __device__ inline v16bf frag_from_f32(const float* __restrict__ p) {
  v8f lo = *(const v8f*)(p);
  v8f hi = *(const v8f*)(p + 16);
  v16bf a;
#pragma unroll
  for (int i = 0; i < 8; ++i) { a[i] = (__bf16)lo[i]; a[i + 8] = (__bf16)hi[i]; }
  return a;
}

static __device__ inline v16bf frag_from_bf16(const __bf16* __restrict__ p) {
  v8bf lo = *(const v8bf*)(p);
  v8bf hi = *(const v8bf*)(p + 16);
  v16bf a;
#pragma unroll
  for (int i = 0; i < 8; ++i) { a[i] = lo[i]; a[i + 8] = hi[i]; }
  return a;
}

template <bool BF>
static __device__ inline v16bf frag_any(const void* __restrict__ base, size_t off) {
  if constexpr (BF) return frag_from_bf16((const __bf16*)base + off);
  else              return frag_from_f32 ((const float*)base + off);
}

// ---------------------------------------------------------------------------
// fp32 -> bf16 bulk conversion (8 elements / thread, b128 in, b128 out)
// ---------------------------------------------------------------------------
__global__ __launch_bounds__(256)
void k_cvt_bf16(const float* __restrict__ src, __bf16* __restrict__ dst, int n8)
{
  const int i = blockIdx.x * 256 + threadIdx.x;
  if (i < n8) {
    v8f v = *(const v8f*)(src + (size_t)i * 8);
    v8bf o;
#pragma unroll
    for (int e = 0; e < 8; ++e) o[e] = (__bf16)v[e];
    *(v8bf*)(dst + (size_t)i * 8) = o;
  }
}

// ---------------------------------------------------------------------------
// Kernel 1: x[b] = GELU(spikes[b] @ W_embed[date[b]]^T + b_embed) * 32 -> bf16
// Wave = 32M x 64N (8 accumulators, B-frags reused across 2 A-frags).
// Block = 4 waves = 32M x 256N. K = 768.
// BF: spikes and W_embed pre-converted to bf16 (conversion-free inner loop).
// ---------------------------------------------------------------------------
template <bool BF>
__global__ __launch_bounds__(128)
void k_embed_gelu(const void*  __restrict__ Sp,
                  const int*   __restrict__ date_idx,
                  const void*  __restrict__ Wemb,
                  const float* __restrict__ b_embed,
                  __bf16*      __restrict__ xws)
{
  const int lane = threadIdx.x & 31;
  const int wave = threadIdx.x >> 5;
  const int b    = blockIdx.z;
  const int t0   = blockIdx.x * 32;
  const int n0   = blockIdx.y * 256 + wave * 64;
  const int mn   = lane & 15;
  const int h    = lane >> 4;
  const int date = date_idx[b];

  const size_t aoff0 = (size_t)(b * TT + t0 + mn) * CC + h * 8;
  const size_t aoff1 = aoff0 + (size_t)16 * CC;
  size_t boff[4];
#pragma unroll
  for (int j = 0; j < 4; ++j)
    boff[j] = (size_t)date * DIN * CC + (size_t)(n0 + j * 16 + mn) * CC + h * 8;

  v8f acc[8] = {{}, {}, {}, {}, {}, {}, {}, {}};

  for (int k0 = 0; k0 < CC; k0 += 32) {
    v16bf a0 = frag_any<BF>(Sp, aoff0 + k0);
    v16bf a1 = frag_any<BF>(Sp, aoff1 + k0);
#pragma unroll
    for (int j = 0; j < 4; ++j) {
      v16bf bf = frag_any<BF>(Wemb, boff[j] + k0);
      acc[j]     = __builtin_amdgcn_wmma_f32_16x16x32_bf16(
                       false, a0, false, bf, (short)0, acc[j],     false, false);
      acc[4 + j] = __builtin_amdgcn_wmma_f32_16x16x32_bf16(
                       false, a1, false, bf, (short)0, acc[4 + j], false, false);
    }
  }

  const float* bias = b_embed + (size_t)date * DIN;
#pragma unroll
  for (int mi = 0; mi < 2; ++mi) {
#pragma unroll
    for (int j = 0; j < 4; ++j) {
      const int col = n0 + j * 16 + mn;
      const float bc = bias[col];
#pragma unroll
      for (int r = 0; r < 8; ++r) {
        const int row = t0 + mi * 16 + h * 8 + r;
        float v = acc[mi * 4 + j][r] + bc;
        float g = 0.5f * v * (1.0f + erff(v * 0.70710678118654752f)) * 32.0f;
        xws[(size_t)(b * TT + row) * DIN + col] = (__bf16)g;
      }
    }
  }
}

// ---------------------------------------------------------------------------
// Kernel 2: out[b, 2+l, :] = stacked[b,l,:] @ W_stack^T + b_stack + pos_table
// stacked row l == contiguous 12288 bf16 at xws row 2l. M=255 (partial),
// N=1024, K=12288. Wave = 32M x 64N. OOB M rows: address-clamped only — a
// garbage A row contaminates only the same (never-stored) C row.
// BF: W_stack pre-converted to bf16.
// ---------------------------------------------------------------------------
template <bool BF>
__global__ __launch_bounds__(128)
void k_stack_gemm(const __bf16* __restrict__ xws,
                  const void*   __restrict__ Wst,
                  const float*  __restrict__ b_stack,
                  const float*  __restrict__ pos_table,
                  const int*    __restrict__ tstamp,
                  float*        __restrict__ out_x)
{
  const int lane = threadIdx.x & 31;
  const int wave = threadIdx.x >> 5;
  const int b    = blockIdx.z;
  const int l0   = blockIdx.x * 32;
  const int n0   = blockIdx.y * 256 + wave * 64;
  const int mn   = lane & 15;
  const int h    = lane >> 4;

  const int la = l0 + mn;
  const int lb = l0 + 16 + mn;
  const int lac = la >= LOUT ? (LOUT - 1) : la;   // clamp: keep reads in-bounds
  const int lbc = lb >= LOUT ? (LOUT - 1) : lb;
  const __bf16* Arow0 = xws + (size_t)(b * TT + 2 * lac) * DIN + h * 8;
  const __bf16* Arow1 = xws + (size_t)(b * TT + 2 * lbc) * DIN + h * 8;

  size_t boff[4];
#pragma unroll
  for (int j = 0; j < 4; ++j)
    boff[j] = (size_t)(n0 + j * 16 + mn) * KSTACK + h * 8;

  v8f acc[8] = {{}, {}, {}, {}, {}, {}, {}, {}};

  for (int k0 = 0; k0 < KSTACK; k0 += 32) {
    v16bf a0 = frag_from_bf16(Arow0 + k0);
    v16bf a1 = frag_from_bf16(Arow1 + k0);
#pragma unroll
    for (int j = 0; j < 4; ++j) {
      v16bf bf = frag_any<BF>(Wst, boff[j] + k0);
      acc[j]     = __builtin_amdgcn_wmma_f32_16x16x32_bf16(
                       false, a0, false, bf, (short)0, acc[j],     false, false);
      acc[4 + j] = __builtin_amdgcn_wmma_f32_16x16x32_bf16(
                       false, a1, false, bf, (short)0, acc[4 + j], false, false);
    }
  }

#pragma unroll
  for (int mi = 0; mi < 2; ++mi) {
#pragma unroll
    for (int j = 0; j < 4; ++j) {
      const int col = n0 + j * 16 + mn;
      const float bs = b_stack[col];
#pragma unroll
      for (int r = 0; r < 8; ++r) {
        const int rl = l0 + mi * 16 + h * 8 + r;
        if (rl < LOUT) {
          const int t = tstamp[b * TT + rl];
          float v = acc[mi * 4 + j][r] + bs + pos_table[(size_t)t * HID + col];
          out_x[((size_t)b * SEQ + 2 + rl) * HID + col] = v;
        }
      }
    }
  }
}

// ---------------------------------------------------------------------------
// Kernel 3: rows 0 (session) / 1 (prompt) per batch + mask/ts outputs.
// final: mask = [1, 1, prod(win)]; ts = [0, 1, orig+2]
// ---------------------------------------------------------------------------
__global__ __launch_bounds__(256)
void k_head(const float* __restrict__ session_table,
            const float* __restrict__ prompt_table,
            const int*   __restrict__ session_idx,
            const int*   __restrict__ mask_idx,
            const int*   __restrict__ spikes_mask,
            const int*   __restrict__ tstamp,
            float*       __restrict__ out)
{
  const int b  = blockIdx.x;
  const int si = session_idx[0];
  const int mi = mask_idx[0];
  float* out_mask = out + (size_t)BB * SEQ * HID;
  float* out_ts   = out_mask + BB * SEQ;

  for (int hcol = threadIdx.x; hcol < HID; hcol += 256) {
    out[((size_t)b * SEQ + 0) * HID + hcol] = session_table[si * HID + hcol];
    out[((size_t)b * SEQ + 1) * HID + hcol] = prompt_table[mi * HID + hcol];
  }
  for (int j = threadIdx.x; j < SEQ; j += 256) {
    float mv, tv;
    if (j < 2) {
      mv = 1.0f; tv = (float)j;
    } else {
      const int l = j - 2;
      const int p = spikes_mask[b * TT + 2 * l] * spikes_mask[b * TT + 2 * l + 1] *
                    spikes_mask[b * TT + 2 * l + 2] * spikes_mask[b * TT + 2 * l + 3];
      mv = (float)p;
      tv = (float)(tstamp[b * TT + l] + 2);
    }
    out_mask[b * SEQ + j] = mv;
    out_ts[b * SEQ + j]   = tv;
  }
}

extern "C" void kernel_launch(void* const* d_in, const int* in_sizes, int n_in,
                              void* d_out, int out_size, void* d_ws, size_t ws_size,
                              hipStream_t stream)
{
  const float* spikes        = (const float*)d_in[0];
  const int*   spikes_mask   = (const int*)d_in[1];
  const int*   spikes_ts     = (const int*)d_in[2];
  const int*   date_idx      = (const int*)d_in[3];
  const int*   mask_idx      = (const int*)d_in[4];
  const int*   session_idx   = (const int*)d_in[5];
  const float* W_embed       = (const float*)d_in[6];
  const float* b_embed       = (const float*)d_in[7];
  const float* W_stack       = (const float*)d_in[8];
  const float* b_stack       = (const float*)d_in[9];
  const float* pos_table     = (const float*)d_in[10];
  const float* prompt_table  = (const float*)d_in[11];
  const float* session_table = (const float*)d_in[12];

  float* out = (float*)d_out;
  char*  ws  = (char*)d_ws;

  const size_t xws_bytes  = (size_t)BB * TT * DIN * sizeof(__bf16);  // 50.3 MB
  const size_t wstk_elems = (size_t)HID * KSTACK;                    // 12.6 M
  const size_t wstk_bytes = wstk_elems * sizeof(__bf16);             // 25.2 MB
  const size_t wemb_elems = (size_t)NDATES * DIN * CC;               // 37.7 M
  const size_t wemb_bytes = wemb_elems * sizeof(__bf16);             // 75.5 MB
  const size_t spk_elems  = (size_t)BB * TT * CC;                    //  6.3 M
  const size_t spk_bytes  = spk_elems * sizeof(__bf16);              // 12.6 MB

  __bf16* xws  = (__bf16*)ws;
  __bf16* wstk = (__bf16*)(ws + xws_bytes);
  __bf16* wemb = (__bf16*)(ws + xws_bytes + wstk_bytes);
  __bf16* spkb = (__bf16*)(ws + xws_bytes + wstk_bytes + wemb_bytes);

  const bool have_wstk = ws_size >= xws_bytes + wstk_bytes;
  const bool have_emb  = ws_size >= xws_bytes + wstk_bytes + wemb_bytes + spk_bytes;

  const dim3 blk(128, 1, 1);
  const dim3 cblk(256, 1, 1);

  // --- pre-convert operands to bf16 (conversion-free GEMM inner loops)
  if (have_wstk) {
    const int n8 = (int)(wstk_elems / 8);
    k_cvt_bf16<<<dim3((n8 + 255) / 256, 1, 1), cblk, 0, stream>>>(W_stack, wstk, n8);
  }
  if (have_emb) {
    const int n8w = (int)(wemb_elems / 8);
    k_cvt_bf16<<<dim3((n8w + 255) / 256, 1, 1), cblk, 0, stream>>>(W_embed, wemb, n8w);
    const int n8s = (int)(spk_elems / 8);
    k_cvt_bf16<<<dim3((n8s + 255) / 256, 1, 1), cblk, 0, stream>>>(spikes, spkb, n8s);
  }

  // --- GEMM1: (B,512,768) x (768,3072) + GELU*32 -> bf16 workspace
  const dim3 g1(TT / 32, DIN / 256, BB);          // 16 x 12 x 16
  if (have_emb)
    k_embed_gelu<true><<<g1, blk, 0, stream>>>(spkb, date_idx, wemb, b_embed, xws);
  else
    k_embed_gelu<false><<<g1, blk, 0, stream>>>(spikes, date_idx, W_embed,
                                                b_embed, xws);

  // --- GEMM2: (B,255,12288) x (12288,1024) + bias + pos_table
  const dim3 g2((LOUT + 31) / 32, HID / 256, BB); // 8 x 4 x 16
  if (have_wstk)
    k_stack_gemm<true><<<g2, blk, 0, stream>>>(xws, wstk, b_stack, pos_table,
                                               spikes_ts, out);
  else
    k_stack_gemm<false><<<g2, blk, 0, stream>>>(xws, W_stack, b_stack,
                                                pos_table, spikes_ts, out);

  // --- prepend rows + mask/ts
  k_head<<<dim3(BB, 1, 1), cblk, 0, stream>>>(
      session_table, prompt_table, session_idx, mask_idx,
      spikes_mask, spikes_ts, out);
}
